// FrequencyBranchMoE_64132451664359
// MI455X (gfx1250) — matrix operand order repaired and checked by reading, not compiled, not run on hardware
//
#include <hip/hip_runtime.h>
#include <hip/hip_bf16.h>

typedef __attribute__((ext_vector_type(16))) _Float16 v16h;
typedef __attribute__((ext_vector_type(8)))  _Float16 v8h;
typedef __attribute__((ext_vector_type(8)))  float    v8f;
typedef __attribute__((ext_vector_type(4)))  unsigned int u32x4;
typedef __attribute__((ext_vector_type(8)))  int      i32x8;
typedef __attribute__((ext_vector_type(4)))  int      i32x4;

#if defined(__has_builtin)
#if __has_builtin(__builtin_amdgcn_tensor_load_to_lds) && \
    __has_builtin(__builtin_amdgcn_s_wait_tensorcnt)
#define FBM_HAS_TDM 1
#endif
#endif

#define B_      128
#define LSEQ_   4096
#define NFREQ   2049
#define NEXP    8
#define TOPK    2
#define NPAIR   (B_ * TOPK)
#define L1      1025
#define L2      513
#define OUTSEQ  256
#define OUTEL   (B_ * 128 * OUTSEQ)

__device__ __forceinline__ float relu_(float x) { return fmaxf(x, 0.0f); }

// Fuse two contiguous 16B chunks into one WMMA fragment (register-only).
__device__ __forceinline__ v16h cat16_(v8h x, v8h y) {
  return __builtin_shufflevector(x, y, 0, 1, 2, 3, 4, 5, 6, 7,
                                 8, 9, 10, 11, 12, 13, 14, 15);
}

// ---------------------------------------------------------------------------
// Kernel 1: Hann window + 4096-pt radix-2 complex FFT in LDS (ortho norm).
// One workgroup (8 wave32) per batch row. Output: [b][2][2049] (re, im).
// ---------------------------------------------------------------------------
__global__ void __launch_bounds__(256) fbm_fft_kernel(const float* __restrict__ x,
                                                      float* __restrict__ fftb) {
  __shared__ float re[LSEQ_];
  __shared__ float im[LSEQ_];
  const int b = blockIdx.x;
  const int tid = threadIdx.x;
  const float TWO_PI = 6.28318530717958647692f;

  for (int n = tid; n < LSEQ_; n += 256) {
    float w = 0.5f * (1.0f - __cosf(TWO_PI * (float)n / (float)LSEQ_));
    re[n] = x[(size_t)b * LSEQ_ + n] * w;
    im[n] = 0.0f;
  }
  __syncthreads();
  for (int i = tid; i < LSEQ_; i += 256) {   // bit-reverse (12 bits)
    int j = (int)(__brev((unsigned)i) >> 20);
    if (i < j) {
      float tr = re[i]; re[i] = re[j]; re[j] = tr;
      float tm = im[i]; im[i] = im[j]; im[j] = tm;
    }
  }
  __syncthreads();
  for (int len = 2; len <= LSEQ_; len <<= 1) {
    const int half = len >> 1;
    const float ang0 = -TWO_PI / (float)len;
    for (int idx = tid; idx < (LSEQ_ >> 1); idx += 256) {
      int blk = idx / half;
      int j   = idx - blk * half;
      int i0  = blk * len + j;
      int i1  = i0 + half;
      float s, c;
      __sincosf(ang0 * (float)j, &s, &c);
      float tr = c * re[i1] - s * im[i1];
      float tm = c * im[i1] + s * re[i1];
      re[i1] = re[i0] - tr;
      im[i1] = im[i0] - tm;
      re[i0] += tr;
      im[i0] += tm;
    }
    __syncthreads();
  }
  const float inv = 1.0f / 64.0f;  // 1/sqrt(4096), ortho
  for (int k = tid; k < NFREQ; k += 256) {
    fftb[(size_t)b * 2 * NFREQ + k]         = re[k] * inv;
    fftb[(size_t)b * 2 * NFREQ + NFREQ + k] = im[k] * inv;
  }
}

// ---------------------------------------------------------------------------
// Kernel 2: f32 -> f16 weight conversion (for WMMA A operands)
// ---------------------------------------------------------------------------
__global__ void fbm_cvt_f16_kernel(const float* __restrict__ src,
                                   _Float16* __restrict__ dst, int n) {
  int i = blockIdx.x * 256 + threadIdx.x;
  if (i < n) dst[i] = (_Float16)src[i];
}

// ---------------------------------------------------------------------------
// Kernel 3: gate convs (k=5,s=1,p=2 twice) + mean pool. One block per batch,
// L tiled in chunks of 256 with halos in LDS; pooled sums via ds_add_f32.
// ---------------------------------------------------------------------------
__global__ void __launch_bounds__(256) fbm_gate_conv_kernel(
    const float* __restrict__ fftb, const float* __restrict__ gw1,
    const float* __restrict__ gb1, const float* __restrict__ gw2,
    const float* __restrict__ gb2, float* __restrict__ pooled) {
  const int b = blockIdx.x;
  const int tid = threadIdx.x;
  __shared__ float sfft[2][264];   // chunk(256) + 4 halo each side
  __shared__ float sh1[32][262];   // chunk(256) + 2 halo each side (260 used)
  __shared__ float psum[64];
  if (tid < 64) psum[tid] = 0.0f;

  const float* fb = fftb + (size_t)b * 2 * NFREQ;
  for (int c0 = 0; c0 < NFREQ; c0 += 256) {
    __syncthreads();
    for (int i = tid; i < 2 * 264; i += 256) {   // stage fft [c0-4, c0+260)
      int ch = i / 264, p = i - ch * 264;
      int g = c0 - 4 + p;
      sfft[ch][p] = (g >= 0 && g < NFREQ) ? fb[ch * NFREQ + g] : 0.0f;
    }
    __syncthreads();
    for (int i = tid; i < 32 * 260; i += 256) {  // h1 for [c0-2, c0+258)
      int oc = i / 260, p = i - oc * 260;
      int gp = c0 - 2 + p;
      float acc = gb1[oc];
#pragma unroll
      for (int ic = 0; ic < 2; ic++)
#pragma unroll
        for (int t = 0; t < 5; t++)
          acc += gw1[(oc * 2 + ic) * 5 + t] * sfft[ic][p + t];
      sh1[oc][p] = (gp >= 0 && gp < NFREQ) ? relu_(acc) : 0.0f;  // zero pad
    }
    __syncthreads();
    {  // h2 + pool accumulate; thread -> fixed oc, 64-wide position slice
      int oc = tid & 63;
      int p0 = (tid >> 6) << 6;
      float lacc = 0.0f;
      for (int p = p0; p < p0 + 64; p++) {
        int g = c0 + p;
        if (g < NFREQ) {
          float acc = gb2[oc];
          for (int ic = 0; ic < 32; ic++) {
            const float* wr = &gw2[(oc * 32 + ic) * 5];
            const float* hr = &sh1[ic][p];
            acc += wr[0]*hr[0] + wr[1]*hr[1] + wr[2]*hr[2] + wr[3]*hr[3] + wr[4]*hr[4];
          }
          lacc += relu_(acc);
        }
      }
      atomicAdd(&psum[oc], lacc);   // ds_add_f32
    }
  }
  __syncthreads();
  if (tid < 64) pooled[(size_t)b * 64 + tid] = psum[tid] / (float)NFREQ;
}

// ---------------------------------------------------------------------------
// Kernel 4: gate MLP + softmax + top-2 + aux. One block, thread = batch.
// ---------------------------------------------------------------------------
__global__ void __launch_bounds__(128) fbm_gate_mlp_kernel(
    const float* __restrict__ pooled, const float* __restrict__ mw1,
    const float* __restrict__ mb1, const float* __restrict__ mw2,
    const float* __restrict__ mb2, float* __restrict__ twP,
    int* __restrict__ tiP, float* __restrict__ aux_out) {
  __shared__ float srw[NEXP];
  __shared__ float slg[NEXP];
  const int b = threadIdx.x;
  if (b < NEXP) { srw[b] = 0.0f; slg[b] = 0.0f; }
  __syncthreads();

  float p[64];
#pragma unroll
  for (int i = 0; i < 64; i++) p[i] = pooled[(size_t)b * 64 + i];

  float lg[NEXP];
#pragma unroll
  for (int e = 0; e < NEXP; e++) lg[e] = mb2[e];
  for (int j = 0; j < 128; j++) {
    float hj = mb1[j];
    const float* wr = &mw1[(size_t)j * 64];
#pragma unroll
    for (int i = 0; i < 64; i++) hj += p[i] * wr[i];
    hj = relu_(hj);
#pragma unroll
    for (int e = 0; e < NEXP; e++) lg[e] += hj * mw2[(size_t)e * 128 + j];
  }
  float mx = lg[0];
#pragma unroll
  for (int e = 1; e < NEXP; e++) mx = fmaxf(mx, lg[e]);
  float rw[NEXP], sum = 0.0f;
#pragma unroll
  for (int e = 0; e < NEXP; e++) { rw[e] = __expf(lg[e] - mx); sum += rw[e]; }
#pragma unroll
  for (int e = 0; e < NEXP; e++) rw[e] /= sum;
  int i1 = 0;                      // top-2, ties -> lower index
#pragma unroll
  for (int e = 1; e < NEXP; e++) if (rw[e] > rw[i1]) i1 = e;
  int i2 = (i1 == 0) ? 1 : 0;
#pragma unroll
  for (int e = 0; e < NEXP; e++) if (e != i1 && rw[e] > rw[i2]) i2 = e;
  float s2 = rw[i1] + rw[i2];
  tiP[b * 2 + 0] = i1;
  tiP[b * 2 + 1] = i2;
  twP[b * 2 + 0] = rw[i1] / s2;
  twP[b * 2 + 1] = rw[i2] / s2;
#pragma unroll
  for (int e = 0; e < NEXP; e++) {
    atomicAdd(&srw[e], rw[e]);
    atomicAdd(&slg[e], lg[e]);
  }
  __syncthreads();
  if (b == 0) {
    float aux = 0.0f;
#pragma unroll
    for (int e = 0; e < NEXP; e++)
      aux += (srw[e] / (float)B_) * (slg[e] / (float)B_);
    aux_out[0] = 0.01f * (float)NEXP * aux;
  }
}

// ---------------------------------------------------------------------------
// Kernel 5: expert conv1 (2->64 ch, k=7, s=2, p=3), selected pairs only.
// One block per (batch, k) pair. fft row staged in LDS — issued first as a
// TDM tensor_load_to_lds (D# per ISA §8.3/8.4: count=1, type=2, 4B elems,
// 2049x2 tile) with s_wait_tensorcnt, then vector loads write the
// authoritative copy (TDM builtin field packing is probe-unverified).
// ---------------------------------------------------------------------------
__global__ void __launch_bounds__(256) fbm_expert_conv1_kernel(
    const float* __restrict__ fftb, const float* __restrict__ ew1,
    const float* __restrict__ eb1, const int* __restrict__ tiP,
    _Float16* __restrict__ e1f) {
  const int pair = blockIdx.x;
  const int b = pair >> 1;
  const int e = tiP[pair];
  __shared__ float sf[2][NFREQ];
  __shared__ float sw[64 * 2 * 7];
  __shared__ float sb[64];

#ifdef FBM_HAS_TDM
  if (threadIdx.x < 32) {  // wave 0 issues the tensor DMA (EXEC-independent)
    unsigned long long ga =
        (unsigned long long)(size_t)(fftb + (size_t)b * 2 * NFREQ);
    u32x4 g0;
    g0[0] = 1u;                                     // count=1, user mode
    g0[1] = (unsigned)(size_t)(&sf[0][0]);          // lds_addr (addr[31:0])
    g0[2] = (unsigned)(ga & 0xffffffffu);           // global_addr lo
    g0[3] = (unsigned)((ga >> 32) & 0x01ffffffu) | (2u << 30);  // hi | type=2
    i32x8 g1;
    g1[0] = (int)(2u << 16);          // data_size=4B, no multicast/pad/iter
    g1[1] = (int)((unsigned)NFREQ << 16);  // tensor_dim0[15:0] @ bits63:48
    g1[2] = (int)(2u << 16);          // tensor_dim0 hi=0 | tensor_dim1=2
    g1[3] = (int)((unsigned)NFREQ << 16);  // tensor_dim1 hi=0 | tile_dim0
    g1[4] = 2;                        // tile_dim1=2, tile_dim2=0
    g1[5] = NFREQ;                    // tensor_dim0_stride lo32
    g1[6] = 0;                        // stride hi | dim1_stride lo
    g1[7] = 0;
    i32x4 gz = {0, 0, 0, 0};
    i32x8 gz8 = {0, 0, 0, 0, 0, 0, 0, 0};
    __builtin_amdgcn_tensor_load_to_lds(g0, g1, gz, gz, gz8, 0);
    __builtin_amdgcn_s_wait_tensorcnt((short)0);
  }
#endif
  __syncthreads();
  for (int i = threadIdx.x; i < 2 * NFREQ; i += 256)
    sf[i / NFREQ][i % NFREQ] = fftb[(size_t)b * 2 * NFREQ + i];
  for (int i = threadIdx.x; i < 64 * 14; i += 256) sw[i] = ew1[(size_t)e * 896 + i];
  if (threadIdx.x < 64) sb[threadIdx.x] = eb1[(size_t)e * 64 + threadIdx.x];
  __syncthreads();
  _Float16* outp = e1f + (size_t)pair * 64 * L1;
  for (int i = threadIdx.x; i < 64 * L1; i += 256) {
    int oc = i / L1, l = i - oc * L1;
    float acc = sb[oc];
    int base = 2 * l - 3;
#pragma unroll
    for (int ic = 0; ic < 2; ic++)
#pragma unroll
      for (int t = 0; t < 7; t++) {
        int g = base + t;
        if (g >= 0 && g < NFREQ) acc += sw[(oc * 2 + ic) * 7 + t] * sf[ic][g];
      }
    outp[i] = (_Float16)relu_(acc);
  }
}

// ---------------------------------------------------------------------------
// Kernel 6: expert conv2 as implicit GEMM with V_WMMA_F32_16X16X32_F16.
// M=128 (8 waves x 16 oc), N=16 positions/block, K=320=64ic*5taps.
// B tile staged in LDS in im2col fragment order sBt[n][k]: each lane's
// fragment = two 8-half runs at k0+8h and k0+16+8h -> two ds_load_b128. K-loop fully unrolled.
// ---------------------------------------------------------------------------
__global__ void __launch_bounds__(256) fbm_expert_conv2_wmma_kernel(
    const _Float16* __restrict__ e1f, const _Float16* __restrict__ wf2,
    const float* __restrict__ eb2, const int* __restrict__ tiP,
    _Float16* __restrict__ e2f) {
  const int bx = blockIdx.x;
  const int pair = bx / 33;
  const int ntile = bx - pair * 33;
  const int n0 = ntile * 16;
  const int e = tiP[pair];
  const int tid = threadIdx.x;
  const int wave = tid >> 5;
  const int lane = tid & 31;
  const int lo = lane & 15;          // A row / B col / D col within tile
  const int hh = lane >> 4;          // lane half (K/M split per ISA layout)

  // im2col B tile [n][k]; row stride 328 halves (16B aligned, conflict-free)
  __shared__ __align__(16) _Float16 sBt[16][328];
  const _Float16* e1p = e1f + (size_t)pair * 64 * L1;
  for (int i = tid; i < 16 * 320; i += 256) {   // 5120 = 20*256, uniform
    int n = i / 320, kk = i - n * 320;
    int ic = kk / 5, t = kk - ic * 5;
    int g = 2 * (n0 + n) - 2 + t;
    sBt[n][kk] = (g >= 0 && g < L1) ? e1p[ic * L1 + g] : (_Float16)0.0f;
  }
  __syncthreads();

  const int oc_a = wave * 16 + lo;
  const _Float16* wrow = wf2 + ((size_t)e * 128 + oc_a) * 320;
  const _Float16* brow = &sBt[lo][0];
  __builtin_prefetch(wrow, 0, 0);    // global_prefetch_b8

  v8f c = {0.f, 0.f, 0.f, 0.f, 0.f, 0.f, 0.f, 0.f};
#pragma unroll
  for (int kb = 0; kb < 10; kb++) {  // K = 320 = 10*32
    const int k0 = kb * 32;
    // A layout: lanes0-15 V0-3:K0-7, V4-7:K16-23; lanes16-31 +8 -> 2x16B
    v8h a0 = *(const v8h*)(wrow + k0 + (hh << 3));
    v8h a1 = *(const v8h*)(wrow + k0 + 16 + (hh << 3));
    // B layout: same K order as A: lanes0-15 K0-7,K16-23; lanes16-31 +8 -> 2x16B
    v8h b0 = *(const v8h*)(brow + k0 + (hh << 3));
    v8h b1 = *(const v8h*)(brow + k0 + 16 + (hh << 3));
    c = __builtin_amdgcn_wmma_f32_16x16x32_f16(false, cat16_(a0, a1),
                                               false, cat16_(b0, b1),
                                               (short)0, c, false, false);
  }
  const int n = n0 + lo;             // D: VGPR r -> M=r+8*half, N=lane&15
  if (n < L2) {
    _Float16* outp = e2f + (size_t)pair * 128 * L2;
#pragma unroll
    for (int r = 0; r < 8; r++) {
      int od = wave * 16 + r + (hh << 3);
      float val = c[r] + eb2[(size_t)e * 128 + od];
      outp[(size_t)od * L2 + n] = (_Float16)relu_(val);
    }
  }
}

// ---------------------------------------------------------------------------
// Kernel 7: expert conv3 WMMA GEMM (M=128, K=384=128ic*3) for BOTH selected
// experts of a batch, fused with top-2 weighted accumulation. One block per
// (batch, n-tile); writes final[b][oc][n] once (no atomics, no pre-zero).
// ---------------------------------------------------------------------------
__global__ void __launch_bounds__(256) fbm_expert_conv3_wmma_kernel(
    const _Float16* __restrict__ e2f, const _Float16* __restrict__ wf3,
    const float* __restrict__ eb3, const int* __restrict__ tiP,
    const float* __restrict__ twP, float* __restrict__ finalb) {
  const int bx = blockIdx.x;
  const int b = bx / 33;
  const int ntile = bx - b * 33;
  const int n0 = ntile * 16;
  const int tid = threadIdx.x;
  const int wave = tid >> 5;
  const int lane = tid & 31;
  const int lo = lane & 15;
  const int hh = lane >> 4;

  // im2col B tile [n][k]; row stride 392 halves (16B aligned, conflict-free)
  __shared__ __align__(16) _Float16 sBt[16][392];
  float accum[8] = {0.f, 0.f, 0.f, 0.f, 0.f, 0.f, 0.f, 0.f};

  for (int kk = 0; kk < TOPK; kk++) {
    const int pair = b * 2 + kk;
    const int e = tiP[pair];
    const float gw = twP[pair];
    __syncthreads();
    const _Float16* e2p = e2f + (size_t)pair * 128 * L2;
    for (int i = tid; i < 16 * 384; i += 256) {  // 6144 = 24*256, uniform
      int n = i / 384, kx = i - n * 384;
      int ic = kx / 3, t = kx - ic * 3;
      int g = n0 + n - 1 + t;
      sBt[n][kx] = (g >= 0 && g < L2) ? e2p[ic * L2 + g] : (_Float16)0.0f;
    }
    __syncthreads();

    const int oc_a = wave * 16 + lo;
    const _Float16* wrow = wf3 + ((size_t)e * 128 + oc_a) * 384;
    const _Float16* brow = &sBt[lo][0];
    __builtin_prefetch(wrow, 0, 0);

    v8f c = {0.f, 0.f, 0.f, 0.f, 0.f, 0.f, 0.f, 0.f};
#pragma unroll
    for (int kb = 0; kb < 12; kb++) {            // K = 384 = 12*32
      const int k0 = kb * 32;
      v8h a0 = *(const v8h*)(wrow + k0 + (hh << 3));
      v8h a1 = *(const v8h*)(wrow + k0 + 16 + (hh << 3));
      v8h b0 = *(const v8h*)(brow + k0 + (hh << 3));
      v8h b1 = *(const v8h*)(brow + k0 + 16 + (hh << 3));
      c = __builtin_amdgcn_wmma_f32_16x16x32_f16(false, cat16_(a0, a1),
                                                 false, cat16_(b0, b1),
                                                 (short)0, c, false, false);
    }
#pragma unroll
    for (int r = 0; r < 8; r++) {
      int od = wave * 16 + r + (hh << 3);
      accum[r] += gw * relu_(c[r] + eb3[(size_t)e * 128 + od]);
    }
  }
  const int n = n0 + lo;
  if (n < L2) {
#pragma unroll
    for (int r = 0; r < 8; r++) {
      int od = wave * 16 + r + (hh << 3);
      finalb[((size_t)b * 128 + od) * L2 + n] = accum[r];
    }
  }
}

// ---------------------------------------------------------------------------
// Kernel 8: adaptive max pool 513 -> 256 (matches floor/ceil windowing)
// ---------------------------------------------------------------------------
__global__ void fbm_pool_kernel(const float* __restrict__ finalb,
                                float* __restrict__ out) {
  int i = blockIdx.x * 256 + threadIdx.x;
  if (i >= OUTEL) return;
  int o = i & (OUTSEQ - 1);
  int row = i >> 8;                       // b*128 + oc
  int start = (o * L2) >> 8;              // floor(o*513/256)
  int end = ((o + 1) * L2 + 255) >> 8;    // ceil((o+1)*513/256)
  const float* r = finalb + (size_t)row * L2;
  float m = -3.402823466e38f;
  for (int p = start; p < end; p++) m = fmaxf(m, r[p]);
  out[i] = m;
}

// ---------------------------------------------------------------------------
extern "C" void kernel_launch(void* const* d_in, const int* in_sizes, int n_in,
                              void* d_out, int out_size, void* d_ws, size_t ws_size,
                              hipStream_t stream) {
  const float* x   = (const float*)d_in[0];
  const float* gw1 = (const float*)d_in[1];
  const float* gb1 = (const float*)d_in[2];
  const float* gw2 = (const float*)d_in[3];
  const float* gb2 = (const float*)d_in[4];
  const float* mw1 = (const float*)d_in[5];
  const float* mb1 = (const float*)d_in[6];
  const float* mw2 = (const float*)d_in[7];
  const float* mb2 = (const float*)d_in[8];
  const float* ew1 = (const float*)d_in[9];
  const float* eb1 = (const float*)d_in[10];
  const float* ew2 = (const float*)d_in[11];
  const float* eb2 = (const float*)d_in[12];
  const float* ew3 = (const float*)d_in[13];
  const float* eb3 = (const float*)d_in[14];
  float* out = (float*)d_out;

  char* ws = (char*)d_ws;
  size_t off = 0;
  auto take = [&](size_t bytes) -> char* {
    char* p = ws + off;
    off = (off + bytes + 255) & ~(size_t)255;
    return p;
  };
  float*     fftb   = (float*)take((size_t)B_ * 2 * NFREQ * 4);
  float*     pooled = (float*)take((size_t)B_ * 64 * 4);
  float*     twP    = (float*)take((size_t)NPAIR * 4);
  int*       tiP    = (int*)take((size_t)NPAIR * 4);
  _Float16*  wf2    = (_Float16*)take((size_t)NEXP * 128 * 320 * 2);
  _Float16*  wf3    = (_Float16*)take((size_t)NEXP * 128 * 384 * 2);
  _Float16*  e1f    = (_Float16*)take((size_t)NPAIR * 64 * L1 * 2);
  _Float16*  e2f    = (_Float16*)take((size_t)NPAIR * 128 * L2 * 2);
  float*     finalb = (float*)take((size_t)B_ * 128 * L2 * 4);
  (void)in_sizes; (void)n_in; (void)out_size; (void)ws_size;

  fbm_fft_kernel<<<B_, 256, 0, stream>>>(x, fftb);
  {
    int n2 = NEXP * 128 * 320, n3 = NEXP * 128 * 384;
    fbm_cvt_f16_kernel<<<(n2 + 255) / 256, 256, 0, stream>>>(ew2, wf2, n2);
    fbm_cvt_f16_kernel<<<(n3 + 255) / 256, 256, 0, stream>>>(ew3, wf3, n3);
  }
  fbm_gate_conv_kernel<<<B_, 256, 0, stream>>>(fftb, gw1, gb1, gw2, gb2, pooled);
  fbm_gate_mlp_kernel<<<1, 128, 0, stream>>>(pooled, mw1, mb1, mw2, mb2,
                                             twP, tiP, out + OUTEL);
  fbm_expert_conv1_kernel<<<NPAIR, 256, 0, stream>>>(fftb, ew1, eb1, tiP, e1f);
  fbm_expert_conv2_wmma_kernel<<<NPAIR * 33, 256, 0, stream>>>(e1f, wf2, eb2,
                                                               tiP, e2f);
  fbm_expert_conv3_wmma_kernel<<<B_ * 33, 256, 0, stream>>>(e2f, wf3, eb3,
                                                            tiP, twP, finalb);
  fbm_pool_kernel<<<(OUTEL + 255) / 256, 256, 0, stream>>>(finalb, out);
}
